// BlockGRUCell_36867999269238
// MI455X (gfx1250) — matrix operand order repaired and checked by reading, not compiled, not run on hardware
//
#include <hip/hip_runtime.h>
#include <hip/hip_bf16.h>

#define GRU_BATCH    16384
#define GRU_HID      1024
#define GRU_INFEATS  2048
#define GRU_NBLK     16
#define GRU_INPER    128
#define GRU_OUTPER   192
#define GRU_OUTFEATS 3072

typedef __attribute__((ext_vector_type(16))) __bf16 v16bf;
typedef __attribute__((ext_vector_type(8)))  float  v8f;

__device__ __forceinline__ unsigned pack_bf16x2(float a, float b) {
    // round-to-nearest-even f32 -> bf16, packed low|high
    union { float f; unsigned u; } ua, ub;
    ua.f = a; ub.f = b;
    unsigned lo = ua.u + (0x7FFFu + ((ua.u >> 16) & 1u));
    unsigned hi = ub.u + (0x7FFFu + ((ub.u >> 16) & 1u));
    return (lo >> 16) | (hi & 0xFFFF0000u);
}

__device__ __forceinline__ float fast_sigmoid(float z) {
    return 1.0f / (1.0f + __expf(-z));
}

// ---------------------------------------------------------------------------
// Kernel 1: repack weights f32 -> bf16 in B-operand register layout.
// Chunk index = ((g*4 + kk)*12 + nb)*32 + lane ; each chunk = 16 bf16 (32 B)
// = W[g][nb*16 + ln][kk*32 + hi*16 + 0..15]  (ln=lane&15, hi=lane>>4),
// this code's contiguous-k 32x16 B order, same as A (lanes 0-15: K 0..15, lanes 16-31: K 16..31).
// ---------------------------------------------------------------------------
__global__ __launch_bounds__(256)
void repack_weights_kernel(const float* __restrict__ w, unsigned* __restrict__ ws)
{
    int id = blockIdx.x * 256 + threadIdx.x;      // 0 .. 24575
    int lane = id & 31;
    int t    = id >> 5;
    int nb   = t % 12;  t /= 12;
    int kk   = t % 4;   t /= 4;
    int g    = t;
    int ln = lane & 15, hi = lane >> 4;

    const float* src = w + ((size_t)g * GRU_OUTPER + nb * 16 + ln) * GRU_INPER
                         + kk * 32 + hi * 16;
    unsigned pk[8];
    #pragma unroll
    for (int q = 0; q < 4; ++q) {
        float4 f = *(const float4*)(src + q * 4);
        pk[q * 2]     = pack_bf16x2(f.x, f.y);
        pk[q * 2 + 1] = pack_bf16x2(f.z, f.w);
    }
    uint4* dst = (uint4*)(ws + (size_t)id * 8);
    dst[0] = make_uint4(pk[0], pk[1], pk[2], pk[3]);
    dst[1] = make_uint4(pk[4], pk[5], pk[6], pk[7]);
}

// ---------------------------------------------------------------------------
// Kernel 2: fused block-GEMM (WMMA bf16) + GRU pointwise.
// One workgroup = 16 batch rows x all 3072 gate cols; wave w owns block w.
// LDS = 64 KB hx(bf16) + 96 KB gate(bf16) = 160 KB -> 2 workgroups per WGP.
// ---------------------------------------------------------------------------
__global__ __launch_bounds__(512)
void block_gru_wmma_kernel(const float* __restrict__ x,
                           const float* __restrict__ h,
                           const unsigned* __restrict__ wbf,  // repacked bf16
                           const float* __restrict__ bias,
                           float* __restrict__ out)
{
    __shared__ unsigned       hx_lds[16 * 1024];     // 16 x 2048 bf16 pairs
    __shared__ __hip_bfloat16 gate_lds[16 * 3072];   // gate pre-activations

    const int tid = threadIdx.x;
    const int b0  = blockIdx.x * 16;

    // ---- Stage hx = [h | x] tile into LDS as bf16 pairs ----
    #pragma unroll 4
    for (int i = 0; i < 32; ++i) {
        int p   = tid + i * 512;             // pair index 0..16383
        int row = p >> 10;
        int cp  = p & 1023;
        int col = cp * 2;
        const float* src = (col < GRU_HID)
            ? (h + (size_t)(b0 + row) * GRU_HID + col)
            : (x + (size_t)(b0 + row) * GRU_HID + (col - GRU_HID));
        float2 v = *(const float2*)src;
        hx_lds[p] = pack_bf16x2(v.x, v.y);
    }
    __syncthreads();

    const int wv   = tid >> 5;               // wave id == GRU block id
    const int lane = tid & 31;
    const int ln   = lane & 15;
    const int hi   = lane >> 4;

    v8f acc[12];
    #pragma unroll
    for (int nb = 0; nb < 12; ++nb) acc[nb] = (v8f)0.0f;

    const unsigned* arow = hx_lds + ln * 1024;
    // flat tile index t = kk*12+nb; chunk(t) = (wv*48 + t)*32 + lane
    const uint4* wp = (const uint4*)wbf + ((size_t)wv * 48 * 32 + lane) * 2;

    union AV { v16bf v; uint4 q[2]; };
    union BV { v16bf v; uint4 q[2]; };

    AV A;
    // preload A for kk = 0 and B for t = 0 (software pipeline prologue)
    {
        int pb = wv * 64 + hi * 8;
        A.q[0] = *(const uint4*)(arow + pb);
        A.q[1] = *(const uint4*)(arow + pb + 4);
    }
    uint4 cur0 = wp[0], cur1 = wp[1];

    #pragma unroll
    for (int t = 0; t < 48; ++t) {
        int kk = t / 12;
        int nb = t - kk * 12;
        if (nb == 0 && t > 0) {              // new K-slab: reload A from LDS
            int pb = wv * 64 + kk * 16 + hi * 8;
            A.q[0] = *(const uint4*)(arow + pb);
            A.q[1] = *(const uint4*)(arow + pb + 4);
        }
        uint4 nxt0, nxt1;
        if (t < 47) {                        // prefetch next tile's B chunk
            nxt0 = wp[(size_t)(t + 1) * 64];
            nxt1 = wp[(size_t)(t + 1) * 64 + 1];
        }
        BV B; B.q[0] = cur0; B.q[1] = cur1;
        acc[nb] = __builtin_amdgcn_wmma_f32_16x16x32_bf16(
            false, A.v, false, B.v, (short)0, acc[nb], false, false);
        if (t < 47) { cur0 = nxt0; cur1 = nxt1; }
    }

    // ---- Spill gate tile (+bias) to LDS as bf16, per 16x16 f32 C/D layout ----
    #pragma unroll
    for (int nb = 0; nb < 12; ++nb) {
        int ncol = wv * GRU_OUTPER + nb * 16 + ln;
        float bv = bias[ncol];
        int rowbase = hi * 8;
        #pragma unroll
        for (int v = 0; v < 8; ++v)
            gate_lds[(rowbase + v) * GRU_OUTFEATS + ncol] =
                __float2bfloat16(acc[nb][v] + bv);
    }
    __syncthreads();

    // ---- Fused GRU pointwise: r/c/u at cols j, 1024+j, 2048+j ----
    #pragma unroll 4
    for (int i = 0; i < 32; ++i) {
        int idx = tid + i * 512;
        int row = idx >> 10;
        int j   = idx & 1023;
        const __hip_bfloat16* g = gate_lds + row * GRU_OUTFEATS;
        float r  = fast_sigmoid(__bfloat162float(g[j]));
        float c  = tanhf(r * __bfloat162float(g[1024 + j]));
        float u  = fast_sigmoid(__bfloat162float(g[2048 + j]));
        float hv = h[(size_t)(b0 + row) * GRU_HID + j];
        out[(size_t)(b0 + row) * GRU_HID + j] = u * c + (1.0f - u) * hv;
    }
}

extern "C" void kernel_launch(void* const* d_in, const int* in_sizes, int n_in,
                              void* d_out, int out_size, void* d_ws, size_t ws_size,
                              hipStream_t stream) {
    const float* x    = (const float*)d_in[0];   // (16384, 1024)
    const float* h    = (const float*)d_in[1];   // (16384, 1024)
    const float* wgt  = (const float*)d_in[2];   // (16, 192, 128)
    const float* bias = (const float*)d_in[3];   // (3072,)
    float* out        = (float*)d_out;           // (16384, 1024)
    unsigned* wbf     = (unsigned*)d_ws;         // 768 KB bf16 repacked weights
    (void)in_sizes; (void)n_in; (void)out_size; (void)ws_size;

    repack_weights_kernel<<<dim3(96), dim3(256), 0, stream>>>(wgt, wbf);
    block_gru_wmma_kernel<<<dim3(GRU_BATCH / 16), dim3(512), 0, stream>>>(
        x, h, wbf, bias, out);
}
